// WindowedMultiHeadAttention_9921374454125
// MI455X (gfx1250) — matrix operand order repaired and checked by reading, not compiled, not run on hardware
//
#include <hip/hip_runtime.h>

// ---------------------------------------------------------------------------
// Swin windowed MHA, fully fused per-window kernel for MI455X (gfx1250).
//   - wave32, 8 waves / workgroup, 1 workgroup per window (4096 windows)
//   - all intermediates (X, Q, K, V^T, logits, O) in LDS (~258 KB of 320 KB WGP LDS)
//   - all GEMMs via v_wmma_f32_16x16x32_f16 (f16 in, f32 accumulate)
//   - weights pre-transposed to N-major f16 so every A and B fragment is a
//     contiguous 16-half run per lane (k = 16*(lane/16) + s, same order for A and B)
//   - B fragments pipelined with prefetch distance 2 (3 live buffers) so the
//     wait before each WMMA quad keeps >=1 younger load pair outstanding
//   - softmax: bias+mask pre-combined; row 1/sum deferred into P@V epilogue
// ---------------------------------------------------------------------------

typedef _Float16 half_t;
typedef half_t v8h  __attribute__((ext_vector_type(8)));
typedef half_t v16h __attribute__((ext_vector_type(16)));
typedef float  v8f  __attribute__((ext_vector_type(8)));

#define WIN_N  49      // tokens per window
#define MPAD   64      // padded token count (4 WMMA M-tiles)
#define CDIM   384     // channels
#define QKV_N  1152    // 3*C
#define NHEAD  12
#define HDIM   32
#define NWG    4
#define SCALEQ 0.17677669529663687f   // 32^-0.5

__device__ __forceinline__ v8f wmma_f16(v16h a, v16h b, v8f c) {
  return __builtin_amdgcn_wmma_f32_16x16x32_f16(false, a, false, b, (short)0, c,
                                                false, false);
}

// A-matrix fragment (16x32 f16), source row-major [M][K] with leading dim ld.
// lane<16: M=lane, K {kb..kb+15};
// lane>=16: M=lane-16, K {kb+16..kb+31}.
__device__ __forceinline__ v16h load_a_frag(const half_t* base, int ld,
                                            int mrow, int kb, int lane) {
  const int hp = lane >> 4;
  const int m  = lane & 15;
  const half_t* p = base + (size_t)(mrow + m) * ld + kb + hp * 16;
  return *(const v16h*)p;
}

// B-matrix fragment (32x16 f16), source stored N-major [N][K] with leading dim
// ld (B[k][n] = base[n*ld + k]).  One contiguous 32B run per lane.
__device__ __forceinline__ v16h load_b_frag(const half_t* base, int ld,
                                            int nbase, int kb, int lane) {
  const int n  = lane & 15;
  const int ko = (lane >> 4) * 16;
  const half_t* p = base + (size_t)(nbase + n) * ld + kb + ko;
  return *(const v16h*)p;
}

// --------------------------- prep kernels ----------------------------------
__global__ void prep_wqkvT(const float* __restrict__ w, half_t* __restrict__ wt) {
  int i = blockIdx.x * 256 + threadIdx.x;           // i = n*384 + k
  if (i < QKV_N * CDIM) {
    int n = i / CDIM;
    int k = i - n * CDIM;
    wt[i] = (half_t)w[k * QKV_N + n];
  }
}

__global__ void prep_wprojT(const float* __restrict__ w, half_t* __restrict__ wt) {
  int i = blockIdx.x * 256 + threadIdx.x;           // i = n*384 + k
  if (i < CDIM * CDIM) {
    int n = i / CDIM;
    int k = i - n * CDIM;
    wt[i] = (half_t)w[k * CDIM + n];
  }
}

// combined additive logit term: biasM[g][h][n][m] = bias_table[rel[n][m]][h] + mask[g][n][m]
__global__ void prep_biasmask(const float* __restrict__ table,
                              const int* __restrict__ ridx,
                              const float* __restrict__ mask,
                              float* __restrict__ biasM) {
  int i = blockIdx.x * 256 + threadIdx.x;   // i = ((g*12)+h)*2401 + nm
  if (i < NWG * NHEAD * WIN_N * WIN_N) {
    int g   = i / (NHEAD * WIN_N * WIN_N);
    int rem = i - g * (NHEAD * WIN_N * WIN_N);
    int h   = rem / (WIN_N * WIN_N);
    int nm  = rem - h * (WIN_N * WIN_N);
    biasM[i] = table[ridx[nm] * NHEAD + h] + mask[g * (WIN_N * WIN_N) + nm];
  }
}

// --------------------------- fused main kernel -----------------------------
__global__ __launch_bounds__(256, 1) void swin_window_attn(
    const float* __restrict__ x, const float* __restrict__ b_qkv,
    const float* __restrict__ b_proj, const half_t* __restrict__ wqkvT,
    const half_t* __restrict__ wprojT, const float* __restrict__ biasM,
    float* __restrict__ out) {
  __shared__ __align__(64) half_t Xs[MPAD * CDIM];     // X f16; reused as O later
  __shared__ __align__(64) half_t Qs[MPAD * CDIM];     // scaled Q, [tok][h*32+d]
  __shared__ __align__(64) half_t Ks[MPAD * CDIM];     // K, [tok][h*32+d]
  __shared__ __align__(64) half_t Vt[CDIM * MPAD];     // V^T, [h*32+d][tok]
  __shared__ __align__(64) half_t Sb[8 * MPAD * MPAD]; // per-wave logits/probs
  __shared__ __align__(64) float  Inv[8 * MPAD];       // per-wave row 1/sum

  const int tid  = threadIdx.x;
  const int wave = tid >> 5;
  const int lane = tid & 31;
  const int col  = lane & 15;
  const int hp   = lane >> 4;
  const int win  = blockIdx.x;
  const int g    = win & (NWG - 1);
  const float* xw = x + (size_t)win * WIN_N * CDIM;
  const v8f zf = {0.f, 0.f, 0.f, 0.f, 0.f, 0.f, 0.f, 0.f};

  // warm the L0/L2 path for this wave's first weight tile (global_prefetch_b8)
  __builtin_prefetch(wqkvT + (size_t)(wave * 9 * 16 + lane) * CDIM, 0, 3);

  // ---- Phase A: stage X into LDS as f16 (pad rows 49..63 with zeros) ----
  for (int i = tid; i < MPAD * CDIM; i += 256) {
    int row = i / CDIM;
    float v = (row < WIN_N) ? xw[i] : 0.f;
    Xs[i] = (half_t)v;
  }
  __syncthreads();

  // ---- Phase B: QKV = X @ Wqkv + b  (72 n-tiles, 9 per wave) ----
  for (int nt = wave * 9; nt < wave * 9 + 9; ++nt) {
    const int nb = nt * 16;
    v8f acc[4] = {zf, zf, zf, zf};
    // prefetch-distance-2 B pipeline: consume b0 while b1 ready, b2 in flight
    v16h b0 = load_b_frag(wqkvT, CDIM, nb, 0, lane);
    v16h b1 = load_b_frag(wqkvT, CDIM, nb, 32, lane);
#pragma unroll
    for (int k = 0; k < 12; ++k) {
      const int kb = k * 32;
      v16h bn = (k + 2 < 12) ? load_b_frag(wqkvT, CDIM, nb, kb + 64, lane) : b0;
#pragma unroll
      for (int m = 0; m < 4; ++m)
        acc[m] = wmma_f16(load_a_frag(Xs, CDIM, m * 16, kb, lane), b0, acc[m]);
      b0 = b1;
      b1 = bn;
    }
    const float bb  = b_qkv[nb + col];
    const int   seg = nb / CDIM;          // 0=Q, 1=K, 2=V (tile never straddles)
    const int   nc  = nb - seg * CDIM;
#pragma unroll
    for (int m = 0; m < 4; ++m) {
#pragma unroll
      for (int r = 0; r < 8; ++r) {
        const int row = m * 16 + r + 8 * hp;
        const float v = acc[m][r] + bb;
        if (seg == 0)      Qs[row * CDIM + nc + col]   = (half_t)(v * SCALEQ);
        else if (seg == 1) Ks[row * CDIM + nc + col]   = (half_t)v;
        else               Vt[(nc + col) * MPAD + row] = (half_t)v;  // transpose
      }
    }
  }
  __syncthreads();

  // ---- Phase C: per-head attention (waves 0..3 take 2 heads, 4..7 take 1) ----
  half_t* S    = Sb + wave * (MPAD * MPAD);
  float*  invw = Inv + wave * MPAD;
  half_t* O    = Xs;  // X is dead; reuse its LDS for attention output
  const int nheads = (wave < NHEAD - 8) ? 2 : 1;
  for (int hh = 0; hh < nheads; ++hh) {
    const int h  = wave + hh * 8;
    const int hb = h * HDIM;
    // logits S = (Q*scale) @ K^T   (K-dim = 32 -> one WMMA per 16x16 tile)
#pragma unroll
    for (int mt = 0; mt < 4; ++mt) {
#pragma unroll
      for (int nt2 = 0; nt2 < 4; ++nt2) {
        v16h a = load_a_frag(Qs, CDIM, mt * 16, hb, lane);
        v16h b = load_b_frag(Ks, CDIM, nt2 * 16, hb, lane);
        v8f acc = wmma_f16(a, b, zf);
#pragma unroll
        for (int r = 0; r < 8; ++r)
          S[(mt * 16 + r + 8 * hp) * MPAD + nt2 * 16 + col] = (half_t)acc[r];
      }
    }
    asm volatile("s_wait_dscnt 0" ::: "memory");  // same-wave LDS RAW fence

    // softmax: 2 rows/lane, 2 passes; normalization deferred to P@V epilogue
    const float* bm = biasM + (size_t)(g * NHEAD + h) * (WIN_N * WIN_N);
#pragma unroll
    for (int rr = 0; rr < 2; ++rr) {
      const int row = lane * 2 + rr;
      half_t* Sr = S + row * MPAD;
      if (row < WIN_N) {
        const float* bmr = bm + row * WIN_N;
        float mx = -1e30f;
        for (int j = 0; j < WIN_N; ++j) {       // pass 1: add bias, find max
          float v = (float)Sr[j] + bmr[j];
          mx = fmaxf(mx, v);
          Sr[j] = (half_t)v;
        }
        float sum = 0.f;
        for (int j = 0; j < WIN_N; ++j) {       // pass 2: exp, accumulate sum
          float e = __expf((float)Sr[j] - mx);
          sum += e;
          Sr[j] = (half_t)e;
        }
        for (int j = WIN_N; j < MPAD; ++j) Sr[j] = (half_t)0.f;  // inert K-pad
        invw[row] = 1.0f / sum;
      } else {
        for (int j = 0; j < MPAD; ++j) Sr[j] = (half_t)0.f;      // inert M-pad
        invw[row] = 0.f;
      }
    }
    asm volatile("s_wait_dscnt 0" ::: "memory");

    // O_h = diag(1/sum) * (E @ V)   (K-dim 64 -> 2 WMMA k-steps)
#pragma unroll
    for (int mt = 0; mt < 4; ++mt) {
#pragma unroll
      for (int dt = 0; dt < 2; ++dt) {
        v8f acc = zf;
        acc = wmma_f16(load_a_frag(S, MPAD, mt * 16, 0, lane),
                       load_b_frag(Vt, MPAD, hb + dt * 16, 0, lane), acc);
        acc = wmma_f16(load_a_frag(S, MPAD, mt * 16, 32, lane),
                       load_b_frag(Vt, MPAD, hb + dt * 16, 32, lane), acc);
#pragma unroll
        for (int r = 0; r < 8; ++r) {
          const int row = mt * 16 + r + 8 * hp;
          O[row * CDIM + hb + dt * 16 + col] = (half_t)(acc[r] * invw[row]);
        }
      }
    }
  }
  __syncthreads();

  // ---- Phase D: out = O @ Wproj + b  (24 n-tiles, 3 per wave) ----
  float* outw = out + (size_t)win * WIN_N * CDIM;
  for (int nt = wave * 3; nt < wave * 3 + 3; ++nt) {
    const int nb = nt * 16;
    v8f acc[4] = {zf, zf, zf, zf};
    v16h b0 = load_b_frag(wprojT, CDIM, nb, 0, lane);
    v16h b1 = load_b_frag(wprojT, CDIM, nb, 32, lane);
#pragma unroll
    for (int k = 0; k < 12; ++k) {
      const int kb = k * 32;
      v16h bn = (k + 2 < 12) ? load_b_frag(wprojT, CDIM, nb, kb + 64, lane) : b0;
#pragma unroll
      for (int m = 0; m < 4; ++m)
        acc[m] = wmma_f16(load_a_frag(O, CDIM, m * 16, kb, lane), b0, acc[m]);
      b0 = b1;
      b1 = bn;
    }
    const float bp = b_proj[nb + col];
#pragma unroll
    for (int m = 0; m < 4; ++m) {
#pragma unroll
      for (int r = 0; r < 8; ++r) {
        const int row = m * 16 + r + 8 * hp;
        if (row < WIN_N) outw[row * CDIM + nb + col] = acc[m][r] + bp;
      }
    }
  }
}

// --------------------------- host launcher ---------------------------------
extern "C" void kernel_launch(void* const* d_in, const int* in_sizes, int n_in,
                              void* d_out, int out_size, void* d_ws,
                              size_t ws_size, hipStream_t stream) {
  const float* x          = (const float*)d_in[0];
  const float* mask       = (const float*)d_in[1];
  const float* w_qkv      = (const float*)d_in[2];
  const float* b_qkv      = (const float*)d_in[3];
  const float* bias_table = (const float*)d_in[4];
  const float* w_proj     = (const float*)d_in[5];
  const float* b_proj     = (const float*)d_in[6];
  const int*   rel_index  = (const int*)d_in[7];
  float* out = (float*)d_out;

  // workspace layout (all offsets 256B aligned):
  //   [0)        wqkvT  f16 [1152][384]      = 884736 B
  //   [884736)   wprojT f16 [384][384]       = 294912 B
  //   [1179648)  biasM  f32 [4][12][49][49]  = 460992 B
  char* ws = (char*)d_ws;
  half_t* wqkvT  = (half_t*)(ws);
  half_t* wprojT = (half_t*)(ws + 884736);
  float*  biasM  = (float*)(ws + 1179648);

  prep_wqkvT<<<(QKV_N * CDIM + 255) / 256, 256, 0, stream>>>(w_qkv, wqkvT);
  prep_wprojT<<<(CDIM * CDIM + 255) / 256, 256, 0, stream>>>(w_proj, wprojT);
  prep_biasmask<<<(NWG * NHEAD * WIN_N * WIN_N + 255) / 256, 256, 0, stream>>>(
      bias_table, rel_index, mask, biasM);

  swin_window_attn<<<4096, 256, 0, stream>>>(x, b_qkv, b_proj, wqkvT, wprojT,
                                             biasM, out);
}
